// RRG_37426345017429
// MI455X (gfx1250) — matrix operand order repaired and checked by reading, not compiled, not run on hardware
//
#include <hip/hip_runtime.h>

// ---------------------------------------------------------------------------
// MI455X (gfx1250) EdgeConv GNN.
//  - per-(b,i) 128x64 GEMMs via V_WMMA_F32_16X16X32_F16 (wave32)
//  - xi-part of the edge MLP folded into a per-node constant (halves K)
//  - ALL LDS staging via CDNA5 async global->LDS B128 copies (ASYNCcnt),
//    sources pre-converted to f16 once per call, so staging is a pure copy
//    that overlaps with the per-node constant compute.
// ---------------------------------------------------------------------------

typedef __attribute__((ext_vector_type(16))) _Float16 v16h;
typedef __attribute__((ext_vector_type(8)))  _Float16 v8h;
typedef __attribute__((ext_vector_type(8)))  float    v8f;

// ------------------------- generic f32 -> f16 copy -------------------------
__global__ void f32_to_f16_kernel(const float* __restrict__ src,
                                  _Float16* __restrict__ dst, int n) {
  int idx = blockIdx.x * 256 + threadIdx.x;
  if (idx < n) dst[idx] = (_Float16)src[idx];
}

// ------------------- weight pre-conversion (GEMM block) --------------------
// Original W is (Din x 64) with rows [0,DX)=W1a (xi), [DX,2DX)=W1b (xj-xi),
// [2DX,2DX+DE)=W1c (edge). GEMM runs on rows [x_j | e_ij] so we emit, per
// output kk slot: kk<DX -> W1b row; [DXp,DXp+DE) -> W1c row; else 0.
// Layout: Wg (64 x KGpad) f16, transposed (column-major K contiguous).
__global__ void convert_wg_kernel(const float* __restrict__ W,
                                  _Float16* __restrict__ Wg,
                                  int DX, int DXp, int DE, int KGpad) {
  int idx = blockIdx.x * 256 + threadIdx.x;
  if (idx >= 64 * KGpad) return;
  int col = idx / KGpad;
  int kk  = idx - col * KGpad;
  float v = 0.0f;
  if (kk < DX)                          v = W[(size_t)(DX + kk) * 64 + col];
  else if (kk >= DXp && kk < DXp + DE)  v = W[(size_t)(2 * DX + (kk - DXp)) * 64 + col];
  Wg[idx] = (_Float16)v;
}

// Wd (64 x DXp) f16: Wd[c][k] = W1a[k,c] - W1b[k,c]  (for the xi constant)
__global__ void convert_wd_kernel(const float* __restrict__ W,
                                  _Float16* __restrict__ Wd,
                                  int DX, int DXp) {
  int idx = blockIdx.x * 256 + threadIdx.x;
  if (idx >= 64 * DXp) return;
  int col = idx / DXp;
  int kk  = idx - col * DXp;
  float v = 0.0f;
  if (kk < DX) v = W[(size_t)kk * 64 + col] - W[(size_t)(DX + kk) * 64 + col];
  Wd[idx] = (_Float16)v;
}

// plain transpose: W (Din x 64) -> Wt (64 x Kfull) f16, zero padded.
__global__ void convert_wt_kernel(const float* __restrict__ W,
                                  _Float16* __restrict__ Wt,
                                  int Din, int Kfull) {
  int idx = blockIdx.x * 256 + threadIdx.x;
  if (idx >= 64 * Kfull) return;
  int col = idx / Kfull;
  int k   = idx - col * Kfull;
  float v = (k < Din) ? W[(size_t)k * 64 + col] : 0.0f;
  Wt[idx] = (_Float16)v;
}

// --------------------------- node embedding --------------------------------
// x0h[b,n,:] = concat( relu(relu(coords@W1+b1)@W2+b2), nf, jt ) as f16,
// row stride 120 (115 real + 5 zero pad so rows are B128-chunkable).
__global__ void node_embed_kernel(const float* __restrict__ coords,
                                  const float* __restrict__ nf,
                                  const float* __restrict__ jt,
                                  const float* __restrict__ W1, const float* __restrict__ b1,
                                  const float* __restrict__ W2, const float* __restrict__ b2,
                                  _Float16* __restrict__ x0h) {
  __shared__ float c3[3];
  __shared__ float h[32];
  int n = blockIdx.x;
  int t = threadIdx.x;                 // 128 threads
  if (t < 3) c3[t] = coords[n * 3 + t];
  __syncthreads();
  if (t < 32) {
    float s = b1[t];
    for (int k = 0; k < 3; ++k) s += c3[k] * W1[k * 32 + t];
    h[t] = fmaxf(s, 0.0f);
  }
  __syncthreads();
  _Float16* row = x0h + (size_t)n * 120;
  if (t < 64) {
    float s = b2[t];
    for (int k = 0; k < 32; ++k) s += h[k] * W2[k * 64 + t];
    row[t] = (_Float16)fmaxf(s, 0.0f);
  } else if (t < 96) {
    row[t] = (_Float16)nf[n * 32 + (t - 64)];
  } else if (t < 115) {
    row[t] = (_Float16)jt[n * 19 + (t - 96)];
  } else if (t < 120) {
    row[t] = (_Float16)0.0f;
  }
}

// --------------------------- small 64->64 relu dense -----------------------
__global__ void dense64_relu_kernel(const float* __restrict__ X,
                                    const float* __restrict__ W,
                                    const float* __restrict__ bb,
                                    float* __restrict__ Y,
                                    _Float16* __restrict__ Yh) {
  __shared__ float xs[64];
  int n = blockIdx.x, t = threadIdx.x;   // 64 threads
  xs[t] = X[(size_t)n * 64 + t];
  __syncthreads();
  float s = bb[t];
  for (int k = 0; k < 64; ++k) s += xs[k] * W[k * 64 + t];
  s = fmaxf(s, 0.0f);
  Y[(size_t)n * 64 + t]  = s;
  Yh[(size_t)n * 64 + t] = (_Float16)s;
}

// --------------------------- WMMA fragment loaders -------------------------
// A 16x32 f16: lane L -> row L%16, k = (L/16)*16 + e, 16 contiguous halves
// (two 16B LDS loads).
__device__ __forceinline__ v16h load_afrag(const _Float16* p) {
  v8h lo = *(const v8h*)p;
  v8h hi = *(const v8h*)(p + 8);
  v16h r;
#pragma unroll
  for (int e = 0; e < 8; ++e) { r[e] = lo[e]; r[e + 8] = hi[e]; }
  return r;
}
// B 32x16 f16: lane L -> col L%16, k = (L/16)*16 + e, 16 contiguous halves.
__device__ __forceinline__ v16h load_bfrag(const _Float16* p) {
  v8h lo = *(const v8h*)p;
  v8h hi = *(const v8h*)(p + 8);
  v16h r;
#pragma unroll
  for (int e = 0; e < 8; ++e) { r[e] = lo[e]; r[e + 8] = hi[e]; }
  return r;
}

// ------------------------- async global->LDS copy --------------------------
__device__ __forceinline__ unsigned lds_off32(const void* p) {
  return (unsigned)(uintptr_t)(const __attribute__((address_space(3))) char*)p;
}
__device__ __forceinline__ void async_b128(unsigned lds_byte, const _Float16* g) {
  asm volatile("global_load_async_to_lds_b128 %0, %1, off"
               :: "v"(lds_byte), "v"((unsigned long long)(uintptr_t)g)
               : "memory");
}
__device__ __forceinline__ void wait_async0() {
  asm volatile("s_wait_asynccnt 0" ::: "memory");
}

// --------------------------- EdgeConv kernel -------------------------------
// One workgroup per (b, i). 8 waves, wave w owns rows j in [16w,16w+16).
//   h1[j] = relu( [x_j | e_ij] @ Wg  +  c_i )     (c_i folds xi and b1)
//   h2[j] = relu( h1[j] @ W2 + b2 )
//   out[i] = max_j ( adj[i,j]>0 ? h2[j] : -1e9 )
__global__ void __launch_bounds__(256) edgeconv_kernel(
    const _Float16* __restrict__ X1h, int DX1, int DX1p,
    const _Float16* __restrict__ X2h, int DX2, int DX2p,
    const _Float16* __restrict__ Eh,  int DE,
    const float* __restrict__ adj,
    const _Float16* __restrict__ Wg, const _Float16* __restrict__ Wd,
    const float* __restrict__ b1v,
    const _Float16* __restrict__ W2t, const float* __restrict__ b2v,
    float* __restrict__ OUT, _Float16* __restrict__ OUTh,
    int KGpad) {
  const int DX   = DX1 + DX2;
  const int DXp  = DX1p + DX2p;
  const int KROW = DXp + DE;                      // staged halves per row
  const int sK   = KGpad + 8;                     // LDS stride (halves)
  extern __shared__ char smem[];
  _Float16* Xs  = (_Float16*)smem;                // 128 x sK
  _Float16* W1s = Xs  + 128 * sK;                 // 64  x sK
  _Float16* W2s = W1s + 64  * sK;                 // 64  x 72
  _Float16* H1s = W2s + 64  * 72;                 // 128 x 72
  float*    xis = (float*)(H1s + 128 * 72);       // 128 f32
  float*    cs  = xis + 128;                      // 64 f32
  float*    red = cs + 64;                        // 8 x 64 f32

  const int bi = blockIdx.x;        // b*128 + i
  const int b  = bi >> 7;
  const int i  = bi & 127;
  const int t  = threadIdx.x;

  const unsigned XsOff  = lds_off32(Xs);
  const unsigned W1sOff = lds_off32(W1s);
  const unsigned W2sOff = lds_off32(W2s);

  // ---- async stage X rows [x_j | e_ij] (B128 chunks, all f16 sources) ----
  {
    const int cpr = KROW >> 3;                    // chunks per row
    const int total = 128 * cpr;
    for (int cid = t; cid < total; cid += 256) {
      int j  = cid / cpr;
      int kk = (cid - j * cpr) << 3;
      const _Float16* src;
      if (kk < DX1p)      src = X1h + (size_t)(b * 128 + j) * DX1p + kk;
      else if (kk < DXp)  src = X2h + (size_t)(b * 128 + j) * DX2p + (kk - DX1p);
      else                src = Eh + ((size_t)bi * 128 + j) * DE + (kk - DXp);
      async_b128(XsOff + (unsigned)(j * sK + kk) * 2u, src);
    }
  }
  // ---- async stage W1 gemm block and W2 ----
  {
    const int cpr = KGpad >> 3;
    for (int cid = t; cid < 64 * cpr; cid += 256) {
      int col = cid / cpr;
      int kk  = (cid - col * cpr) << 3;
      async_b128(W1sOff + (unsigned)(col * sK + kk) * 2u,
                 Wg + (size_t)col * KGpad + kk);
    }
    for (int cid = t; cid < 64 * 8; cid += 256) {
      int col = cid >> 3, kk = (cid & 7) << 3;
      async_b128(W2sOff + (unsigned)(col * 72 + kk) * 2u,
                 W2t + col * 64 + kk);
    }
  }
  // ---- zero-fill the K pad tail of Xs; stage x_i (f32) ----
  if (KROW < KGpad) {
    const int padw = KGpad - KROW;
    for (int idx = t; idx < 128 * padw; idx += 256) {
      int j = idx / padw, r = idx - j * padw;
      Xs[j * sK + KROW + r] = (_Float16)0.0f;
    }
  }
  for (int k = t; k < DX; k += 256) {
    xis[k] = (k < DX1) ? (float)X1h[(size_t)(b * 128 + i) * DX1p + k]
                       : (float)X2h[(size_t)(b * 128 + i) * DX2p + (k - DX1)];
  }
  __syncthreads();
  // ---- per-node constant (overlaps the in-flight async copies) ----
  if (t < 64) {
    const _Float16* wc = Wd + (size_t)t * DXp;
    float s = b1v[t];
    for (int k = 0; k < DX; ++k) s += xis[k] * (float)wc[k];
    cs[t] = s;
  }
  wait_async0();
  __syncthreads();

  const int lane = t & 31;
  const int w    = t >> 5;          // wave id 0..7
  const int lrow = lane & 15;
  const int lhi  = lane >> 4;
  const int rowA = (w << 4) | lrow;

  // ================= GEMM1: H1 = relu(Xs @ W1 + cs) =================
  v8f acc[4];
#pragma unroll
  for (int tt = 0; tt < 4; ++tt)
#pragma unroll
    for (int e = 0; e < 8; ++e) acc[tt][e] = 0.0f;

  for (int ks = 0; ks < KGpad; ks += 32) {
    v16h a = load_afrag(Xs + rowA * sK + ks + lhi * 16);
#pragma unroll
    for (int tt = 0; tt < 4; ++tt) {
      v16h bf = load_bfrag(W1s + (((tt << 4) | lrow)) * sK + ks + lhi * 16);
      acc[tt] = __builtin_amdgcn_wmma_f32_16x16x32_f16(
          false, a, false, bf, (short)0, acc[tt], false, false);
    }
  }
#pragma unroll
  for (int tt = 0; tt < 4; ++tt) {
    int col = (tt << 4) | lrow;
    float cc = cs[col];
#pragma unroll
    for (int r = 0; r < 8; ++r) {
      int row = (w << 4) + (lhi << 3) + r;      // D layout: M = r + 8*lhi
      H1s[row * 72 + col] = (_Float16)fmaxf(acc[tt][r] + cc, 0.0f);
    }
  }
  __syncthreads();

  // ================= GEMM2: H2 = relu(H1 @ W2 + b2) =================
  v8f ac2[4];
#pragma unroll
  for (int tt = 0; tt < 4; ++tt)
#pragma unroll
    for (int e = 0; e < 8; ++e) ac2[tt][e] = 0.0f;

#pragma unroll
  for (int ks = 0; ks < 64; ks += 32) {
    v16h a = load_afrag(H1s + rowA * 72 + ks + lhi * 16);
#pragma unroll
    for (int tt = 0; tt < 4; ++tt) {
      v16h bf = load_bfrag(W2s + (((tt << 4) | lrow)) * 72 + ks + lhi * 16);
      ac2[tt] = __builtin_amdgcn_wmma_f32_16x16x32_f16(
          false, a, false, bf, (short)0, ac2[tt], false, false);
    }
  }

  // ---- adjacency-masked max over j ----
  const float* adjrow = adj + (size_t)bi * 128;
  float av[8];
#pragma unroll
  for (int r = 0; r < 8; ++r) av[r] = adjrow[(w << 4) + (lhi << 3) + r];

#pragma unroll
  for (int tt = 0; tt < 4; ++tt) {
    int col = (tt << 4) | lrow;
    float bb = b2v[col];
    float pm = -1.0e9f;
#pragma unroll
    for (int r = 0; r < 8; ++r) {
      float v = fmaxf(ac2[tt][r] + bb, 0.0f);
      v = (av[r] > 0.0f) ? v : -1.0e9f;
      pm = fmaxf(pm, v);
    }
    pm = fmaxf(pm, __shfl_xor(pm, 16, 32));     // combine the two row-halves
    if (lhi == 0) red[w * 64 + col] = pm;
  }
  __syncthreads();
  if (t < 64) {
    float m = red[t];
#pragma unroll
    for (int ww = 1; ww < 8; ++ww) m = fmaxf(m, red[ww * 64 + t]);
    OUT[(size_t)bi * 64 + t]  = m;
    OUTh[(size_t)bi * 64 + t] = (_Float16)m;
  }
}

// --------------------------- output heads ----------------------------------
// x = [ec3 | ec2];  x1 = relu(relu(x@W4+b4)@Wo1+bo1);  x2 analog with W5/Wo2.
__global__ void __launch_bounds__(256) head_kernel(
    const float* __restrict__ A, const float* __restrict__ B2,
    const float* __restrict__ W4, const float* __restrict__ b4,
    const float* __restrict__ Wo1, const float* __restrict__ bo1,
    const float* __restrict__ W5, const float* __restrict__ b5,
    const float* __restrict__ Wo2, const float* __restrict__ bo2,
    float* __restrict__ out1, float* __restrict__ out2) {
  __shared__ float x[128];
  __shared__ float y1[128];
  __shared__ float y2[128];
  int n = blockIdx.x;
  int t = threadIdx.x;                 // 256 threads
  if (t < 64)       x[t] = A[(size_t)n * 64 + t];
  else if (t < 128) x[t] = B2[(size_t)n * 64 + (t - 64)];
  __syncthreads();
  {
    int c = t & 127;
    const float* W = (t < 128) ? W4 : W5;
    float s = (t < 128) ? b4[c] : b5[c];
    for (int k = 0; k < 128; ++k) s += x[k] * W[k * 128 + c];
    s = fmaxf(s, 0.0f);
    if (t < 128) y1[c] = s; else y2[c] = s;
  }
  __syncthreads();
  if (t < 3) {
    float s = bo1[t];
    for (int k = 0; k < 128; ++k) s += y1[k] * Wo1[k * 3 + t];
    out1[(size_t)n * 3 + t] = fmaxf(s, 0.0f);
  } else if (t == 3) {
    float s = bo2[0];
    for (int k = 0; k < 128; ++k) s += y2[k] * Wo2[k];
    out2[n] = fmaxf(s, 0.0f);
  }
}

// --------------------------- host launcher ---------------------------------
extern "C" void kernel_launch(void* const* d_in, const int* in_sizes, int n_in,
                              void* d_out, int out_size, void* d_ws, size_t ws_size,
                              hipStream_t stream) {
  (void)in_sizes; (void)n_in; (void)out_size; (void)ws_size;
  int p = 0;
  const float* coords = (const float*)d_in[p++];
  const float* adj    = (const float*)d_in[p++];
  const float* nf     = (const float*)d_in[p++];
  const float* efeat  = (const float*)d_in[p++];
  const float* jt     = (const float*)d_in[p++];
  // params in insertion order: hid1, hid2, ecE1, ecE2, hid3, ec1, ec2, ec3,
  // hid4, out1, hid5, out2   (each dense = (w,b); each mlp = 2 denses)
  const float* hid1_w = (const float*)d_in[p++]; const float* hid1_b = (const float*)d_in[p++];
  const float* hid2_w = (const float*)d_in[p++]; const float* hid2_b = (const float*)d_in[p++];
  const float* ecE1_w1 = (const float*)d_in[p++]; const float* ecE1_b1 = (const float*)d_in[p++];
  const float* ecE1_w2 = (const float*)d_in[p++]; const float* ecE1_b2 = (const float*)d_in[p++];
  const float* ecE2_w1 = (const float*)d_in[p++]; const float* ecE2_b1 = (const float*)d_in[p++];
  const float* ecE2_w2 = (const float*)d_in[p++]; const float* ecE2_b2 = (const float*)d_in[p++];
  const float* hid3_w = (const float*)d_in[p++]; const float* hid3_b = (const float*)d_in[p++];
  const float* ec1_w1 = (const float*)d_in[p++]; const float* ec1_b1 = (const float*)d_in[p++];
  const float* ec1_w2 = (const float*)d_in[p++]; const float* ec1_b2 = (const float*)d_in[p++];
  const float* ec2_w1 = (const float*)d_in[p++]; const float* ec2_b1 = (const float*)d_in[p++];
  const float* ec2_w2 = (const float*)d_in[p++]; const float* ec2_b2 = (const float*)d_in[p++];
  const float* ec3_w1 = (const float*)d_in[p++]; const float* ec3_b1 = (const float*)d_in[p++];
  const float* ec3_w2 = (const float*)d_in[p++]; const float* ec3_b2 = (const float*)d_in[p++];
  const float* hid4_w = (const float*)d_in[p++]; const float* hid4_b = (const float*)d_in[p++];
  const float* out1_w = (const float*)d_in[p++]; const float* out1_b = (const float*)d_in[p++];
  const float* hid5_w = (const float*)d_in[p++]; const float* hid5_b = (const float*)d_in[p++];
  const float* out2_w = (const float*)d_in[p++]; const float* out2_b = (const float*)d_in[p++];

  // ---- workspace carve-up (256B aligned) ----
  char* ws = (char*)d_ws;
  size_t off = 0;
  auto walloc = [&](size_t bytes) -> char* {
    char* r = ws + off;
    off += (bytes + 255) & ~(size_t)255;
    return r;
  };
  // f32 activations
  float* oE1 = (float*)walloc(2048ull * 64 * 4);
  float* oE2 = (float*)walloc(2048ull * 64 * 4);
  float* oh3 = (float*)walloc(2048ull * 64 * 4);
  float* oc1 = (float*)walloc(2048ull * 64 * 4);
  float* oc2 = (float*)walloc(2048ull * 64 * 4);
  float* oc3 = (float*)walloc(2048ull * 64 * 4);
  // f16 activations (async-copy sources)
  _Float16* x0h  = (_Float16*)walloc(2048ull * 120 * 2);
  _Float16* oE1h = (_Float16*)walloc(2048ull * 64 * 2);
  _Float16* oE2h = (_Float16*)walloc(2048ull * 64 * 2);
  _Float16* oh3h = (_Float16*)walloc(2048ull * 64 * 2);
  _Float16* oc1h = (_Float16*)walloc(2048ull * 64 * 2);
  _Float16* oc2h = (_Float16*)walloc(2048ull * 64 * 2);
  _Float16* oc3h = (_Float16*)walloc(2048ull * 64 * 2);
  _Float16* Eh   = (_Float16*)walloc(4194304ull * 2);      // (16,128,128,16) f16
  // f16 weights
  _Float16* wgE1 = (_Float16*)walloc(64ull * 160 * 2);
  _Float16* wdE1 = (_Float16*)walloc(64ull * 120 * 2);
  _Float16* w2E1 = (_Float16*)walloc(64ull * 64 * 2);
  _Float16* wgE2 = (_Float16*)walloc(64ull * 96 * 2);
  _Float16* wdE2 = (_Float16*)walloc(64ull * 64 * 2);
  _Float16* w2E2 = (_Float16*)walloc(64ull * 64 * 2);
  _Float16* wg1  = (_Float16*)walloc(64ull * 64 * 2);
  _Float16* wd1  = (_Float16*)walloc(64ull * 64 * 2);
  _Float16* w21  = (_Float16*)walloc(64ull * 64 * 2);
  _Float16* wg2  = (_Float16*)walloc(64ull * 64 * 2);
  _Float16* wd2  = (_Float16*)walloc(64ull * 64 * 2);
  _Float16* w22  = (_Float16*)walloc(64ull * 64 * 2);
  _Float16* wg3  = (_Float16*)walloc(64ull * 128 * 2);
  _Float16* wd3  = (_Float16*)walloc(64ull * 128 * 2);
  _Float16* w23  = (_Float16*)walloc(64ull * 64 * 2);

  // allow >64KB dynamic LDS for the edge-conv kernel (WGP has 320KB)
  hipFuncSetAttribute(reinterpret_cast<const void*>(edgeconv_kernel),
                      hipFuncAttributeMaxDynamicSharedMemorySize, 131072);

  // ---- one-time pre-conversions ----
  f32_to_f16_kernel<<<(4194304 + 255) / 256, 256, 0, stream>>>(efeat, Eh, 4194304);
  auto cwg = [&](const float* W, _Float16* Wg, int DX, int DXp, int DE, int KGpad) {
    convert_wg_kernel<<<(64 * KGpad + 255) / 256, 256, 0, stream>>>(W, Wg, DX, DXp, DE, KGpad);
  };
  auto cwd = [&](const float* W, _Float16* Wd, int DX, int DXp) {
    convert_wd_kernel<<<(64 * DXp + 255) / 256, 256, 0, stream>>>(W, Wd, DX, DXp);
  };
  auto cwt = [&](const float* W, _Float16* Wt, int Din, int Kfull) {
    convert_wt_kernel<<<(64 * Kfull + 255) / 256, 256, 0, stream>>>(W, Wt, Din, Kfull);
  };
  cwg(ecE1_w1, wgE1, 115, 120, 16, 160); cwd(ecE1_w1, wdE1, 115, 120); cwt(ecE1_w2, w2E1, 64, 64);
  cwg(ecE2_w1, wgE2,  64,  64, 16,  96); cwd(ecE2_w1, wdE2,  64,  64); cwt(ecE2_w2, w2E2, 64, 64);
  cwg(ec1_w1,  wg1,   64,  64,  0,  64); cwd(ec1_w1,  wd1,   64,  64); cwt(ec1_w2,  w21, 64, 64);
  cwg(ec2_w1,  wg2,   64,  64,  0,  64); cwd(ec2_w1,  wd2,   64,  64); cwt(ec2_w2,  w22, 64, 64);
  cwg(ec3_w1,  wg3,  128, 128,  0, 128); cwd(ec3_w1,  wd3,  128, 128); cwt(ec3_w2,  w23, 64, 64);

  node_embed_kernel<<<2048, 128, 0, stream>>>(coords, nf, jt,
      hid1_w, hid1_b, hid2_w, hid2_b, x0h);

  auto shb = [](int KGpad) -> size_t {
    int sK = KGpad + 8;
    return (size_t)(128 * sK * 2 + 64 * sK * 2 + 64 * 72 * 2 + 128 * 72 * 2 +
                    (128 + 64 + 512) * 4);
  };

  // ecE1: DX=115 (pad 120), DE=16 -> KGpad=160
  edgeconv_kernel<<<2048, 256, shb(160), stream>>>(
      x0h, 115, 120, nullptr, 0, 0, Eh, 16, adj,
      wgE1, wdE1, ecE1_b1, w2E1, ecE1_b2, oE1, oE1h, 160);
  // ecE2: DX=64, DE=16 -> KGpad=96
  edgeconv_kernel<<<2048, 256, shb(96), stream>>>(
      oE1h, 64, 64, nullptr, 0, 0, Eh, 16, adj,
      wgE2, wdE2, ecE2_b1, w2E2, ecE2_b2, oE2, oE2h, 96);

  dense64_relu_kernel<<<2048, 64, 0, stream>>>(oE2, hid3_w, hid3_b, oh3, oh3h);

  // ec1 / ec2: DX=64, DE=0 -> KGpad=64
  edgeconv_kernel<<<2048, 256, shb(64), stream>>>(
      oh3h, 64, 64, nullptr, 0, 0, nullptr, 0, adj,
      wg1, wd1, ec1_b1, w21, ec1_b2, oc1, oc1h, 64);
  edgeconv_kernel<<<2048, 256, shb(64), stream>>>(
      oc1h, 64, 64, nullptr, 0, 0, nullptr, 0, adj,
      wg2, wd2, ec2_b1, w22, ec2_b2, oc2, oc2h, 64);
  // ec3: input = [ec2_out | ec1_out], DX=128 -> KGpad=128
  edgeconv_kernel<<<2048, 256, shb(128), stream>>>(
      oc2h, 64, 64, oc1h, 64, 64, nullptr, 0, adj,
      wg3, wd3, ec3_b1, w23, ec3_b2, oc3, oc3h, 128);

  float* out1 = (float*)d_out;            // (16,128,3)
  float* out2 = out1 + 2048 * 3;          // (16,128,1)
  head_kernel<<<2048, 256, 0, stream>>>(oc3, oc2,
      hid4_w, hid4_b, out1_w, out1_b,
      hid5_w, hid5_b, out2_w, out2_b, out1, out2);
}
